// Attention_74672301408506
// MI455X (gfx1250) — matrix operand order repaired and checked by reading, not compiled, not run on hardware
//
#include <hip/hip_runtime.h>
#include <hip/hip_bf16.h>
#include <stdint.h>

#define T_DIM 4096
#define H_DIM 256
#define QD    256
#define V_DIM 80
#define VB_ROWS (T_DIM + 8)   // 8 zero rows implement shift truncation
#define KSTRIDE 40            // ushorts per staged K row (64B data + 16B pad)
#define L2E 1.44269504088896340736f

typedef __bf16 v16bf __attribute__((ext_vector_type(16)));
typedef float  v8f   __attribute__((ext_vector_type(8)));
typedef unsigned int u32x4 __attribute__((ext_vector_type(4)));

union Frag { v16bf v; u32x4 u[2]; };

__device__ __forceinline__ unsigned short f2bf(float f) {
  unsigned u = __builtin_bit_cast(unsigned, f);
  unsigned r = 0x7FFFu + ((u >> 16) & 1u);      // round-to-nearest-even
  return (unsigned short)((u + r) >> 16);
}

#define WMMA_BF16(a, b, c) \
  __builtin_amdgcn_wmma_f32_16x16x32_bf16(false, (a), false, (b), (short)0, (c), false, false)

// Async DMA: 16B global -> LDS, tracked by ASYNCcnt (CDNA5 §15.18.3 op 98).
// Low 32 bits of a generic pointer into LDS are the LDS byte address (§10.2).
__device__ __forceinline__ void async_ld16(void* lds, const void* g) {
  unsigned l = (unsigned)(uintptr_t)lds;
  asm volatile("global_load_async_to_lds_b128 %0, %1, off" :: "v"(l), "v"(g) : "memory");
}
__device__ __forceinline__ void wait_async0() {
  asm volatile("s_wait_asynccnt 0" ::: "memory");
}

// Stage one 32-key K^T slab: 256 rows x 64B, one row per thread (4 x b128).
__device__ __forceinline__ void stage_K(unsigned short (*buf)[KSTRIDE],
                                        const unsigned short* __restrict__ Ktb,
                                        int k0, int tid) {
  const unsigned short* g = Ktb + (size_t)tid * T_DIM + k0;
#pragma unroll
  for (int j = 0; j < 4; ++j) async_ld16(&buf[tid][j * 8], g + j * 8);
}

// Stage the 40-row V slab (rows k0..k0+39, 160B each = 400 x b128).
__device__ __forceinline__ void stage_V(unsigned short (*buf)[V_DIM],
                                        const unsigned short* __restrict__ Vb,
                                        int k0, int tid) {
#pragma unroll
  for (int j = 0; j < 2; ++j) {
    int s = tid + j * 256;
    if (s < 400) {
      int row = s / 10, part = s - row * 10;
      async_ld16(&buf[row][part * 8], Vb + (size_t)(k0 + row) * V_DIM + part * 8);
    }
  }
}

// ---------------- projection: Out = (X @ W^T) * scale  (bf16, opt. transposed) --
// Q rows are pre-scaled by log2(e) so QK^T lands directly in the exp2 domain.
__global__ void proj_kernel(const float* __restrict__ X, const float* __restrict__ W,
                            unsigned short* __restrict__ Out, int transpose, float scale) {
  int idx = blockIdx.x * blockDim.x + threadIdx.x;   // over T*H
  int t = idx >> 8;
  int h = idx & 255;
  const float* xr = X + t * QD;
  const float* wr = W + h * QD;
  float acc = 0.f;
#pragma unroll 8
  for (int j = 0; j < QD; ++j) acc = fmaf(xr[j], wr[j], acc);
  acc *= scale;
  if (transpose) Out[h * T_DIM + t] = f2bf(acc);
  else           Out[t * H_DIM + h] = f2bf(acc);
}

// ---------------- value conversion f32 -> bf16 with 8 zero pad rows ------------
__global__ void vconv_kernel(const float* __restrict__ Xv, unsigned short* __restrict__ Vb) {
  int idx = blockIdx.x * blockDim.x + threadIdx.x;
  if (idx >= VB_ROWS * V_DIM) return;
  float v = (idx < T_DIM * V_DIM) ? Xv[idx] : 0.f;
  Vb[idx] = f2bf(v);
}

// ---------------- pass 1: softmax denominators L = sum exp --------------------
__global__ __launch_bounds__(256) void stats_kernel(const unsigned short* __restrict__ Qb,
                                                    const unsigned short* __restrict__ Ktb,
                                                    float* __restrict__ Lsum) {
  __shared__ unsigned short Kb[2][H_DIM][KSTRIDE];   // double-buffered K^T slabs
  const int tid  = threadIdx.x;
  const int lane = tid & 31;
  const int wv   = tid >> 5;
  const int q0   = (blockIdx.x * 8 + wv) * 16;
  const int m    = lane & 15;
  const int hi   = lane >> 4;

  // Q A-fragments: lane = M row, K split per ISA 7.12.2
  Frag A[8];
  {
    const unsigned short* qrow = Qb + (size_t)(q0 + m) * H_DIM + hi * 8;
#pragma unroll
    for (int hc = 0; hc < 8; ++hc) {
      const unsigned short* p = qrow + hc * 32;
      A[hc].u[0] = *(const u32x4*)(p);        // K = hc*32 + (hi?8:0) + 0..7
      A[hc].u[1] = *(const u32x4*)(p + 16);   // K = hc*32 + 16 + (hi?8:0) + 0..7
    }
  }

  float sacc[8];
#pragma unroll
  for (int r = 0; r < 8; ++r) sacc[r] = 0.f;

  stage_K(Kb[0], Ktb, 0, tid);
  wait_async0();
  __syncthreads();

  for (int c = 0; c < T_DIM / 32; ++c) {
    if (c + 1 < T_DIM / 32) stage_K(Kb[(c + 1) & 1], Ktb, (c + 1) * 32, tid);
    const unsigned short(*kb)[KSTRIDE] = Kb[c & 1];

    v8f c0 = {}, c1 = {};
#pragma unroll
    for (int hc = 0; hc < 8; ++hc) {
      const unsigned short* p = &kb[hc * 32][m];  // B lane = key column (N), elements = h (K)
      Frag B0, B1;
#pragma unroll
      for (int e = 0; e < 16; ++e) {
        int kk = (e & 7) + hi * 8 + (e >> 3) * 16;
        B0.v[e] = __builtin_bit_cast(__bf16, p[kk * KSTRIDE]);
        B1.v[e] = __builtin_bit_cast(__bf16, p[kk * KSTRIDE + 16]);
      }
      c0 = WMMA_BF16(A[hc].v, B0.v, c0);
      c1 = WMMA_BF16(A[hc].v, B1.v, c1);
    }
#pragma unroll
    for (int r = 0; r < 8; ++r)
      sacc[r] += __builtin_amdgcn_exp2f(c0[r]) + __builtin_amdgcn_exp2f(c1[r]);

    wait_async0();
    __syncthreads();
  }

  // reduce per-lane partials across the 16-lane half owning each row
#pragma unroll
  for (int r = 0; r < 8; ++r) {
    float s = sacc[r];
    s += __shfl_xor(s, 1, 32);
    s += __shfl_xor(s, 2, 32);
    s += __shfl_xor(s, 4, 32);
    s += __shfl_xor(s, 8, 32);
    sacc[r] = s;
  }
  if ((lane & 15) == 0) {
#pragma unroll
    for (int r = 0; r < 8; ++r) Lsum[q0 + hi * 8 + r] = sacc[r];
  }
}

// ---------------- pass 2: P = softmax(QK^T); Y_s = P @ V_shift(2s) -------------
__global__ __launch_bounds__(256) void attn_kernel(const unsigned short* __restrict__ Qb,
                                                   const unsigned short* __restrict__ Ktb,
                                                   const unsigned short* __restrict__ Vb,
                                                   const float* __restrict__ Lsum,
                                                   float* __restrict__ Y) {
  __shared__ unsigned short Kb[2][H_DIM][KSTRIDE];   // 40 KB
  __shared__ unsigned short Vs[2][40][V_DIM];        // 12.5 KB
  __shared__ unsigned short Pl[8][16][32];           // 8 KB, per-wave P bounce
  const int tid  = threadIdx.x;
  const int lane = tid & 31;
  const int wv   = tid >> 5;
  const int q0   = (blockIdx.x * 8 + wv) * 16;
  const int m    = lane & 15;
  const int hi   = lane >> 4;

  Frag A[8];
  {
    const unsigned short* qrow = Qb + (size_t)(q0 + m) * H_DIM + hi * 8;
#pragma unroll
    for (int hc = 0; hc < 8; ++hc) {
      const unsigned short* p = qrow + hc * 32;
      A[hc].u[0] = *(const u32x4*)(p);
      A[hc].u[1] = *(const u32x4*)(p + 16);
    }
  }
  // fold 1/L into the exponent: p = exp2(c - log2(L))
  float lb[8];
#pragma unroll
  for (int r = 0; r < 8; ++r) lb[r] = -__builtin_amdgcn_logf(Lsum[q0 + hi * 8 + r]);

  v8f acc[20];                               // [shift s][v-tile vt] = acc[s*5+vt]
#pragma unroll
  for (int i = 0; i < 20; ++i) acc[i] = (v8f){};

  stage_K(Kb[0], Ktb, 0, tid);
  stage_V(Vs[0], Vb, 0, tid);
  wait_async0();
  __syncthreads();

  for (int c = 0; c < T_DIM / 32; ++c) {
    const int k0 = c * 32;
    if (c + 1 < T_DIM / 32) {
      stage_K(Kb[(c + 1) & 1], Ktb, k0 + 32, tid);
      stage_V(Vs[(c + 1) & 1], Vb, k0 + 32, tid);
    }
    const unsigned short(*kb)[KSTRIDE] = Kb[c & 1];
    const unsigned short(*vs)[V_DIM]   = Vs[c & 1];

    // ---- scores for 32 keys: 16-WMMA chain fed from LDS ----
    v8f c0 = {}, c1 = {};
#pragma unroll
    for (int hc = 0; hc < 8; ++hc) {
      const unsigned short* p = &kb[hc * 32][m];
      Frag B0, B1;
#pragma unroll
      for (int e = 0; e < 16; ++e) {
        int kk = (e & 7) + hi * 8 + (e >> 3) * 16;
        B0.v[e] = __builtin_bit_cast(__bf16, p[kk * KSTRIDE]);
        B1.v[e] = __builtin_bit_cast(__bf16, p[kk * KSTRIDE + 16]);
      }
      c0 = WMMA_BF16(A[hc].v, B0.v, c0);
      c1 = WMMA_BF16(A[hc].v, B1.v, c1);
    }

    // ---- normalized probs, transpose C-layout -> A-layout through LDS ----
#pragma unroll
    for (int r = 0; r < 8; ++r) {
      float p0 = __builtin_amdgcn_exp2f(c0[r] + lb[r]);
      float p1 = __builtin_amdgcn_exp2f(c1[r] + lb[r]);
      Pl[wv][hi * 8 + r][m]      = f2bf(p0);
      Pl[wv][hi * 8 + r][m + 16] = f2bf(p1);
    }
    asm volatile("s_wait_dscnt 0" ::: "memory");   // same-wave cross-lane LDS RAW
    Frag P;
    P.u[0] = *(const u32x4*)(&Pl[wv][m][hi * 8]);        // K = (hi?8:0)+0..7
    P.u[1] = *(const u32x4*)(&Pl[wv][m][16 + hi * 8]);   // K = 16+(hi?8:0)+0..7

    // ---- P tile feeds 4 shifted value fragments x 5 v-tiles ----
#pragma unroll
    for (int s = 0; s < 4; ++s) {
      const unsigned short* vr = &vs[2 * s][m];   // rows k0+2s+K, column m
      Frag Bv[5];
#pragma unroll
      for (int vt = 0; vt < 5; ++vt) {
        const unsigned short* p = vr + vt * 16;
#pragma unroll
        for (int e = 0; e < 16; ++e) {
          int kk = (e & 7) + hi * 8 + (e >> 3) * 16;
          Bv[vt].v[e] = __builtin_bit_cast(__bf16, p[kk * V_DIM]);
        }
      }
#pragma unroll
      for (int vt = 0; vt < 5; ++vt)
        acc[s * 5 + vt] = WMMA_BF16(P.v, Bv[vt].v, acc[s * 5 + vt]);
    }

    wait_async0();
    __syncthreads();
  }

  // ---- write Y_s tiles ----
#pragma unroll
  for (int s = 0; s < 4; ++s)
#pragma unroll
    for (int vt = 0; vt < 5; ++vt)
#pragma unroll
      for (int r = 0; r < 8; ++r)
        Y[(size_t)s * T_DIM * V_DIM + (size_t)(q0 + hi * 8 + r) * V_DIM + vt * 16 + m] =
            acc[s * 5 + vt][r];
}

// ---------------- shift-mix: out[q] = sum_s w[s] * Y_s[q-2s] -------------------
__global__ void combine_kernel(const float* __restrict__ Y, const float* __restrict__ Wsh,
                               float* __restrict__ Out) {
  int idx = blockIdx.x * blockDim.x + threadIdx.x;
  if (idx >= T_DIM * V_DIM) return;
  int q = idx / V_DIM;
  int v = idx - q * V_DIM;
  float o = 0.f;
#pragma unroll
  for (int s = 0; s < 4; ++s) {
    int qs = q - 2 * s;
    if (qs >= 0) o += Wsh[s] * Y[((size_t)s * T_DIM + qs) * V_DIM + v];
  }
  Out[idx] = o;
}

extern "C" void kernel_launch(void* const* d_in, const int* in_sizes, int n_in,
                              void* d_out, int out_size, void* d_ws, size_t ws_size,
                              hipStream_t stream) {
  const float* x   = (const float*)d_in[0];
  const float* xk  = (const float*)d_in[1];
  const float* xv  = (const float*)d_in[2];
  const float* Wqk = (const float*)d_in[3];
  const float* wsh = (const float*)d_in[4];
  float* out = (float*)d_out;

  char* ws = (char*)d_ws;
  unsigned short* Qb   = (unsigned short*)(ws + 0);            // 2,097,152 B (pre-scaled by log2e)
  unsigned short* Ktb  = (unsigned short*)(ws + 2097152);      // 2,097,152 B  [H][T]
  unsigned short* Vb   = (unsigned short*)(ws + 4194304);      //   656,640 B  [T+8][V]
  float*          Lsm  = (float*)         (ws + 4851200);      //    16,384 B
  float*          Y    = (float*)         (ws + 4867584);      // 5,242,880 B  [4][T][V]

  proj_kernel<<<(T_DIM * H_DIM) / 256, 256, 0, stream>>>(x,  Wqk, Qb,  0, L2E);
  proj_kernel<<<(T_DIM * H_DIM) / 256, 256, 0, stream>>>(xk, Wqk, Ktb, 1, 1.0f);
  vconv_kernel<<<(VB_ROWS * V_DIM + 255) / 256, 256, 0, stream>>>(xv, Vb);
  stats_kernel<<<32, 256, 0, stream>>>(Qb, Ktb, Lsm);
  attn_kernel<<<32, 256, 0, stream>>>(Qb, Ktb, Vb, Lsm, Y);
  combine_kernel<<<(T_DIM * V_DIM + 255) / 256, 256, 0, stream>>>(Y, wsh, out);
}
